// LinearOnlyNet_83425444758218
// MI455X (gfx1250) — 8-bit matrix operand order repaired and checked by reading, compiled, not run on hardware
//
#include <hip/hip_runtime.h>
#include <cstdint>
#include <cstddef>


typedef __attribute__((ext_vector_type(8))) int          v8i;
typedef __attribute__((ext_vector_type(4))) int          v4i_t;
typedef __attribute__((ext_vector_type(4))) unsigned int v4u_t;

#if defined(__has_builtin)
#if __has_builtin(__builtin_amdgcn_tensor_load_to_lds) && \
    __has_builtin(__builtin_amdgcn_s_wait_tensorcnt)
#define USE_TDM 1
#endif
#if __has_builtin(__builtin_amdgcn_sched_barrier)
#define SCHED_FENCE() __builtin_amdgcn_sched_barrier(0)
#endif
#endif
#ifndef USE_TDM
#define USE_TDM 0
#endif
#ifndef SCHED_FENCE
#define SCHED_FENCE() ((void)0)
#endif

#define BM 128
#define BN 64
#define BK 64

#if USE_TDM
__device__ __forceinline__ void tdm_tile_load(void* lds_dst, const void* gsrc,
                                              unsigned tensor_d0, unsigned tensor_d1,
                                              unsigned stride_d0,
                                              unsigned tile_d0, unsigned tile_d1)
{
    const unsigned           lds_addr = (unsigned)(size_t)lds_dst;
    const unsigned long long ga       = (unsigned long long)(size_t)gsrc;

    v4u_t g0;
    g0[0] = 1u;
    g0[1] = lds_addr;
    g0[2] = (unsigned)ga;
    g0[3] = (unsigned)((ga >> 32) & 0x01ffffffu)
          | (2u << 30);

    v8i g1;
    g1[0] = 0;
    g1[1] = (int)((tensor_d0 & 0xffffu) << 16);
    g1[2] = (int)((tensor_d0 >> 16) | ((tensor_d1 & 0xffffu) << 16));
    g1[3] = (int)((tensor_d1 >> 16) | (tile_d0 << 16));
    g1[4] = (int)tile_d1;
    g1[5] = (int)stride_d0;
    g1[6] = 0;
    g1[7] = 0;

    v4i_t g2 = {0, 0, 0, 0};
    v4i_t g3 = {0, 0, 0, 0};
    v8i   g4 = {0, 0, 0, 0, 0, 0, 0, 0};
    __builtin_amdgcn_tensor_load_to_lds(g0, g1, g2, g3, g4, 0);
}
#endif

__global__ void __launch_bounds__(256)
quantize_x_u8(const float* __restrict__ x, unsigned* __restrict__ x8,
              const float* __restrict__ p_sx, int n4)
{
    const int i = blockIdx.x * 256 + threadIdx.x;
    if (i >= n4) return;
    const float  inv = 1.0f / p_sx[0];
    const float4 v   = ((const float4*)x)[i];
    const unsigned b0 = (unsigned)(int)rintf(fminf(fmaxf(v.x * inv, 0.f), 255.f));
    const unsigned b1 = (unsigned)(int)rintf(fminf(fmaxf(v.y * inv, 0.f), 255.f));
    const unsigned b2 = (unsigned)(int)rintf(fminf(fmaxf(v.z * inv, 0.f), 255.f));
    const unsigned b3 = (unsigned)(int)rintf(fminf(fmaxf(v.w * inv, 0.f), 255.f));
    x8[i] = b0 | (b1 << 8) | (b2 << 16) | (b3 << 24);
}

__global__ void __launch_bounds__(256)
quantize_w_i8(const float* __restrict__ w, unsigned* __restrict__ w8,
              const float* __restrict__ p_sw, int n4)
{
    const int i = blockIdx.x * 256 + threadIdx.x;
    if (i >= n4) return;
    const float  inv = 1.0f / p_sw[0];
    const float4 v   = ((const float4*)w)[i];
    const unsigned b0 = (unsigned)((int)rintf(fminf(fmaxf(v.x * inv, -8.f), 7.f)) & 0xff);
    const unsigned b1 = (unsigned)((int)rintf(fminf(fmaxf(v.y * inv, -8.f), 7.f)) & 0xff);
    const unsigned b2 = (unsigned)((int)rintf(fminf(fmaxf(v.z * inv, -8.f), 7.f)) & 0xff);
    const unsigned b3 = (unsigned)((int)rintf(fminf(fmaxf(v.w * inv, -8.f), 7.f)) & 0xff);
    w8[i] = b0 | (b1 << 8) | (b2 << 16) | (b3 << 24);
}

__global__ void __launch_bounds__(256)
cim_bitserial_gemm(const unsigned char* __restrict__ x8,
                   const unsigned char* __restrict__ w8,
                   const float* __restrict__ bias,
                   const float* __restrict__ gain,
                   const float* __restrict__ p_sx,
                   const float* __restrict__ p_sw,
                   const float* __restrict__ p_so,
                   float* __restrict__ out,
                   int Brows, int IN, int OUT)
{
    __shared__ unsigned char sA[2][BM * BK];
    __shared__ unsigned char sB[2][BN * BK];

    const int tid  = threadIdx.x;
    const int wave = tid >> 5;
    const int lane = tid & 31;
    const int half = lane >> 4;
    const int l16  = lane & 15;
    const int wm   = wave >> 1;
    const int wn   = wave & 1;
    const int m0   = blockIdx.y * BM;
    const int n0   = blockIdx.x * BN;

    const unsigned char* gA = x8 + (size_t)m0 * IN;
    const unsigned char* gB = w8 + (size_t)n0 * IN;

    v8i acc[4][2][2] = {};

    const int ksteps = IN / BK;

#if USE_TDM
    if (wave == 0) {
        tdm_tile_load(sA[0], gA, (unsigned)IN, (unsigned)Brows, (unsigned)IN, BK, BM);
        tdm_tile_load(sB[0], gB, (unsigned)IN, (unsigned)OUT,   (unsigned)IN, BK, BN);
        __builtin_amdgcn_s_wait_tensorcnt(0);
    }
#else
    for (int c = tid; c < (BM * BK) / 16; c += 256) {
        const int row = c >> 2, off = (c & 3) << 4;
        *(uint4*)(sA[0] + row * BK + off) = *(const uint4*)(gA + (size_t)row * IN + off);
    }
    {
        const int row = tid >> 2, off = (tid & 3) << 4;
        *(uint4*)(sB[0] + row * BK + off) = *(const uint4*)(gB + (size_t)row * IN + off);
    }
#endif
    __syncthreads();

    for (int kt = 0; kt < ksteps; ++kt) {
        const int cur  = kt & 1;
        const int nxt  = cur ^ 1;
        const int knxt = (kt + 1) * BK;
        const bool has_next = (kt + 1) < ksteps;

#if USE_TDM
        if (wave == 0 && has_next) {
            tdm_tile_load(sA[nxt], gA + knxt, (unsigned)IN, (unsigned)Brows, (unsigned)IN, BK, BM);
            tdm_tile_load(sB[nxt], gB + knxt, (unsigned)IN, (unsigned)OUT,   (unsigned)IN, BK, BN);
        }
#else
        if (has_next) {
            for (int c = tid; c < (BM * BK) / 16; c += 256) {
                const int row = c >> 2, off = (c & 3) << 4;
                *(uint4*)(sA[nxt] + row * BK + off) =
                    *(const uint4*)(gA + knxt + (size_t)row * IN + off);
            }
            const int row = tid >> 2, off = (tid & 3) << 4;
            *(uint4*)(sB[nxt] + row * BK + off) =
                *(const uint4*)(gB + knxt + (size_t)row * IN + off);
        }
#endif

        v8i rawA[2], fragB[2];
#pragma unroll
        for (int mt = 0; mt < 2; ++mt) {
            const unsigned char* ab = sA[cur] + (wm * 32 + mt * 16 + l16) * BK + half * 8;
#pragma unroll
            for (int c = 0; c < 4; ++c) {
                const unsigned long long q = *(const unsigned long long*)(ab + c * 16);
                rawA[mt][2 * c]     = (int)(unsigned)q;
                rawA[mt][2 * c + 1] = (int)(unsigned)(q >> 32);
            }
        }
#pragma unroll
        for (int nt = 0; nt < 2; ++nt) {
            const unsigned char* bb = sB[cur] + (wn * 32 + nt * 16 + l16) * BK + half * 8;
#pragma unroll
            for (int c = 0; c < 4; ++c) {
                const unsigned long long q = *(const unsigned long long*)(bb + c * 16);
                fragB[nt][2 * c]     = (int)(unsigned)q;
                fragB[nt][2 * c + 1] = (int)(unsigned)(q >> 32);
            }
        }

        v8i aS[4][2];
#pragma unroll
        for (int s = 0; s < 4; ++s) {
            const int sh = 2 * s;
#pragma unroll
            for (int mt = 0; mt < 2; ++mt)
#pragma unroll
                for (int j = 0; j < 8; ++j)
                    aS[s][mt][j] = (int)((((unsigned)rawA[mt][j]) >> sh) & 0x03030303u);
        }
#pragma unroll
        for (int s = 0; s < 4; ++s) {
            asm volatile("" : "+v"(aS[s][0]), "+v"(aS[s][1]));
        }
        asm volatile("" : "+v"(fragB[0]), "+v"(fragB[1]));
        SCHED_FENCE();

#pragma unroll
        for (int s = 0; s < 4; ++s) {
#pragma unroll
            for (int mt = 0; mt < 2; ++mt) {
                acc[s][mt][0] = __builtin_amdgcn_wmma_i32_16x16x64_iu8(
                    false, aS[s][mt], true, fragB[0], acc[s][mt][0], false, false);
                acc[s][mt][1] = __builtin_amdgcn_wmma_i32_16x16x64_iu8(
                    false, aS[s][mt], true, fragB[1], acc[s][mt][1], false, false);
            }
        }
        SCHED_FENCE();

#if USE_TDM
        if (wave == 0 && has_next)
            __builtin_amdgcn_s_wait_tensorcnt(0);
#endif
        __syncthreads();
    }

    const float s_x  = p_sx[0];
    const float s_w  = p_sw[0];
    const float s_o  = p_so[0];
    const float sxsw = s_x * s_w;
    const float so_inv = 1.0f / s_o;

#pragma unroll
    for (int nt = 0; nt < 2; ++nt) {
        const int   col  = n0 + wn * 32 + nt * 16 + l16;
        const float g    = gain[col];
        const float ginv = 1.0f / g;
        const float b    = bias[col];
#pragma unroll
        for (int mt = 0; mt < 2; ++mt) {
            const int rbase = m0 + wm * 32 + mt * 16 + half * 8;
#pragma unroll
            for (int r = 0; r < 8; ++r) {
                float a = 0.f, sh = 1.f;
#pragma unroll
                for (int s = 0; s < 4; ++s) {
                    const float p   = (float)acc[s][mt][nt][r];
                    const float adc = fminf(fmaxf(rintf(p * g), -128.f), 127.f) * ginv;
                    a = fmaf(adc, sh, a);
                    sh *= 4.f;
                }
                const float o = fmaf(a, sxsw, b);
                const float y = s_o * rintf(fminf(fmaxf(o * so_inv, -128.f), 127.f));
                out[(size_t)(rbase + r) * OUT + col] = y;
            }
        }
    }
}

extern "C" void kernel_launch(void* const* d_in, const int* in_sizes, int n_in,
                              void* d_out, int out_size, void* d_ws, size_t ws_size,
                              hipStream_t stream)
{
    const float* x    = (const float*)d_in[0];
    const float* w    = (const float*)d_in[1];
    const float* bias = (const float*)d_in[2];
    const float* s_x  = (const float*)d_in[3];
    const float* s_w  = (const float*)d_in[4];
    const float* s_o  = (const float*)d_in[5];
    const float* gain = (const float*)d_in[6];

    const int OUT = in_sizes[2];
    const int IN  = in_sizes[1] / OUT;
    const int B   = in_sizes[0] / IN;

    unsigned char* x8 = (unsigned char*)d_ws;
    unsigned char* w8 = x8 + (size_t)B * IN;

    const int n4x = (B * IN) / 4;
    quantize_x_u8<<<(n4x + 255) / 256, 256, 0, stream>>>(x, (unsigned*)x8, s_x, n4x);
    const int n4w = (OUT * IN) / 4;
    quantize_w_i8<<<(n4w + 255) / 256, 256, 0, stream>>>(w, (unsigned*)w8, s_w, n4w);

    dim3 grid(OUT / BN, B / BM);
    cim_bitserial_gemm<<<grid, 256, 0, stream>>>(x8, w8, bias, gain, s_x, s_w, s_o,
                                                 (float*)d_out, B, IN, OUT);
}
